// MultiHeadedAttention_61117384622702
// MI455X (gfx1250) — matrix operand order repaired and checked by reading, not compiled, not run on hardware
//
#include <hip/hip_runtime.h>
#include <hip/hip_bf16.h>
#include <math.h>

// Problem constants (reference: B=2, S=2048, D=768, H=12, DK=64)
#define BATCH  2
#define SEQ    2048
#define DMODEL 768
#define NHEAD  12
#define DHEAD  64
#define NROWS  (BATCH*SEQ)        // 4096
#define DD     (DMODEL*DMODEL)    // 589824
#define ND     (NROWS*DMODEL)     // 3145728
#define NEGVAL (-1000000000.0f)

typedef __attribute__((ext_vector_type(16))) __bf16 v16bf;
typedef __attribute__((ext_vector_type(8)))  __bf16 v8bf;
typedef __attribute__((ext_vector_type(8)))  float  v8f;
typedef int vint4 __attribute__((vector_size(16)));   // matches builtin param

// gfx1250 async global->LDS copies (ASYNCcnt path), with safe fallback.
#if defined(__has_builtin)
#if __has_builtin(__builtin_amdgcn_global_load_async_to_lds_b128)
#define HAVE_ASYNC_LDS 1
#endif
#endif
#ifndef HAVE_ASYNC_LDS
#define HAVE_ASYNC_LDS 0
#endif

__device__ __forceinline__ void async_copy16(const void* g, void* l) {
#if HAVE_ASYNC_LDS
  __builtin_amdgcn_global_load_async_to_lds_b128((vint4*)g, (vint4*)l, 0, 0);
#else
  *(v8bf*)l = *(const v8bf*)g;
#endif
}

// Wait for async copies (if any) then block barrier.
__device__ __forceinline__ void stage_wait() {
#if HAVE_ASYNC_LDS
#if defined(__has_builtin) && __has_builtin(__builtin_amdgcn_s_wait_asynccnt)
  __builtin_amdgcn_s_wait_asynccnt(0);
#else
  asm volatile("s_wait_asynccnt 0" ::: "memory");
#endif
#endif
  __syncthreads();
}

// Wave-local LDS flush: same-wave DS ops are in-order; just drain DScnt.
__device__ __forceinline__ void lds_flush_wave() {
  asm volatile("s_wait_dscnt 0" ::: "memory");
}

__device__ __forceinline__ __bf16 f2bf(float f) {
  union { float f; unsigned u; } x; x.f = f;
  unsigned u = x.u;
  unsigned r = u + 0x7FFFu + ((u >> 16) & 1u);   // round-to-nearest-even
  unsigned short h = (unsigned short)(r >> 16);
  __bf16 b; __builtin_memcpy(&b, &h, 2); return b;
}

__device__ __forceinline__ v16bf cat16(v8bf lo, v8bf hi) {
  return __builtin_shufflevector(lo, hi, 0,1,2,3,4,5,6,7,8,9,10,11,12,13,14,15);
}

// ---------------------------------------------------------------------------
// Kernel 1: convert the four weight matrices fp32 -> bf16 into workspace
// ---------------------------------------------------------------------------
__global__ __launch_bounds__(256)
void k_cvt_w(const float* __restrict__ Wq, const float* __restrict__ Wk,
             const float* __restrict__ Wv, const float* __restrict__ Wo,
             __bf16* __restrict__ dst) {
  const float* src = (blockIdx.y == 0) ? Wq : (blockIdx.y == 1) ? Wk
                   : (blockIdx.y == 2) ? Wv : Wo;
  __bf16* d = dst + (size_t)blockIdx.y * DD;
  int i = (blockIdx.x * 256 + threadIdx.x) * 4;
  if (i < DD) {
    float4 v = *(const float4*)(src + i);
    d[i + 0] = f2bf(v.x); d[i + 1] = f2bf(v.y);
    d[i + 2] = f2bf(v.z); d[i + 3] = f2bf(v.w);
  }
}

// ---------------------------------------------------------------------------
// Kernel 2: QKV projection GEMM.  y = x @ W^T + b, bf16 WMMA, B staged in LDS
//   (shared by all 4 waves).  blockIdx.z: 0=Q, 1=K, 2=V.
//   Block = 4 waves -> 64 rows x 64 cols.  Q/K out [B,H,S,64]; V out [B,H,64,S].
// ---------------------------------------------------------------------------
__global__ __launch_bounds__(128)
void k_proj(const float* __restrict__ Xq, const float* __restrict__ Xk,
            const float* __restrict__ Xv, const __bf16* __restrict__ Wb,
            const float* __restrict__ bq, const float* __restrict__ bk,
            const float* __restrict__ bv,
            __bf16* __restrict__ Qb, __bf16* __restrict__ Kb,
            __bf16* __restrict__ Vt) {
  __shared__ __bf16 Bs[64 * 32] __attribute__((aligned(32)));   // [col][k] 4KB

  const int mode = blockIdx.z;
  const float* X    = (mode == 0) ? Xq : (mode == 1) ? Xk : Xv;
  const __bf16* W   = Wb + (size_t)mode * DD;
  const float* bias = (mode == 0) ? bq : (mode == 1) ? bk : bv;

  const int tid  = threadIdx.x;
  const int wave = tid >> 5;
  const int lane = tid & 31;
  const int nlo  = lane & 15;
  const int half = lane >> 4;
  const int row0 = blockIdx.y * 64 + wave * 16;
  const int col0 = blockIdx.x * 64;

  // staging assignment: each thread copies 32B (one 16-elem K segment)
  const int scol = tid >> 1;              // 0..63
  const int skp  = (tid & 1) * 16;        // 0 or 16
  const __bf16* wsrc = W + (size_t)(col0 + scol) * DMODEL + skp;

  v8f acc[4] = {{}, {}, {}, {}};
  const float* arow = X + (size_t)(row0 + nlo) * DMODEL;

  for (int kb = 0; kb < DMODEL; kb += 32) {
    // stage B chunk (64 cols x 32 k) into LDS via async copies
    async_copy16(wsrc + kb,     &Bs[scol * 32 + skp]);
    async_copy16(wsrc + kb + 8, &Bs[scol * 32 + skp + 8]);
    if (kb + 32 < DMODEL) __builtin_prefetch(wsrc + kb + 32, 0, 0);
    stage_wait();

    // A fragment (16x32 bf16): lane = row M, element s = K 16*half + s (same order as B).
    v16bf a;
    const float* p0 = arow + kb + half * 16;
    const float* p1 = arow + kb + half * 16 + 8;
#pragma unroll
    for (int j = 0; j < 8; ++j) a[j]     = f2bf(p0[j]);
#pragma unroll
    for (int j = 0; j < 8; ++j) a[8 + j] = f2bf(p1[j]);
#pragma unroll
    for (int t = 0; t < 4; ++t) {
      v16bf bf = *(const v16bf*)&Bs[(t * 16 + nlo) * 32 + half * 16];
      acc[t] = __builtin_amdgcn_wmma_f32_16x16x32_bf16(
          false, a, false, bf, (short)0, acc[t], false, false);
    }
    __syncthreads();   // all waves done reading Bs before next staging
  }

#pragma unroll
  for (int t = 0; t < 4; ++t) {
    const int col = col0 + t * 16 + nlo;
    const float bb = bias[col];
    const int hh = col >> 6;       // head
    const int dk = col & 63;       // within-head dim
    const int srow0 = row0 + half * 8;          // first of 8 C rows this lane
    const int b  = srow0 >> 11;                 // /SEQ (2048)
    const int si = srow0 & (SEQ - 1);
    if (mode < 2) {
      __bf16* outp = (mode == 0) ? Qb : Kb;
#pragma unroll
      for (int r = 0; r < 8; ++r)
        outp[((size_t)(b * NHEAD + hh) * SEQ + si + r) * DHEAD + dk] =
            f2bf(acc[t][r] + bb);
    } else {
      __bf16* o = Vt + ((size_t)(b * NHEAD + hh) * DHEAD + dk) * SEQ + si;
#pragma unroll
      for (int r = 0; r < 8; ++r) o[r] = f2bf(acc[t][r] + bb);
    }
  }
}

// ---------------------------------------------------------------------------
// Kernel 3: flash attention.  Block = 4 waves; wave w owns query tile
//   q0 = blockIdx.x*64 + w*16.  K chunk (32x64) and V chunk (64x32) staged in
//   LDS once per block per 32-key step (4x traffic reduction vs per-wave).
// ---------------------------------------------------------------------------
__global__ __launch_bounds__(128)
void k_attn(const __bf16* __restrict__ Qb, const __bf16* __restrict__ Kb,
            const __bf16* __restrict__ Vt, const int* __restrict__ mask,
            __bf16* __restrict__ ctx) {
  __shared__ __bf16 Ks[32 * 64] __attribute__((aligned(32)));     // [key][d]
  __shared__ __bf16 Vs[64 * 32] __attribute__((aligned(32)));     // [d][key]
  __shared__ __bf16 pbuf[4 * 16 * 32] __attribute__((aligned(32)));

  const int tid  = threadIdx.x;
  const int wave = tid >> 5;
  const int lane = tid & 31;
  const int nlo  = lane & 15;
  const int half = lane >> 4;
  const int bh = blockIdx.y;
  const int b  = bh / NHEAD;
  const int h  = bh % NHEAD;
  const int q0 = blockIdx.x * 64 + wave * 16;

  __bf16* pw = pbuf + wave * (16 * 32);

  // Q A-fragments for K-depth 0..31 and 32..63 (registers, whole kernel)
  const __bf16* Qrow = Qb + ((size_t)bh * SEQ + q0 + nlo) * DHEAD;
  v16bf aq0 = cat16(*(const v8bf*)(Qrow + half * 16),
                    *(const v8bf*)(Qrow + 8 + half * 16));
  v16bf aq1 = cat16(*(const v8bf*)(Qrow + 32 + half * 16),
                    *(const v8bf*)(Qrow + 40 + half * 16));

  v8f acc[4] = {{}, {}, {}, {}};
  float mrow[8], lrow[8];
#pragma unroll
  for (int r = 0; r < 8; ++r) { mrow[r] = -INFINITY; lrow[r] = 0.f; }

  const __bf16* Kbase = Kb + (size_t)bh * SEQ * DHEAD;
  const __bf16* Vbase = Vt + (size_t)bh * DHEAD * SEQ;
  const int* mptr = mask + b * SEQ;

  // staging assignments (per thread, 32B each for K and V)
  const int kkey = tid >> 2;             // 0..31
  const int kdp  = (tid & 3) * 16;       // 0,16,32,48
  const int vd   = tid >> 1;             // 0..63
  const int vkp  = (tid & 1) * 16;       // 0,16
  const __bf16* ksrc0 = Kbase + (size_t)kkey * DHEAD + kdp;
  const __bf16* vsrc0 = Vbase + (size_t)vd * SEQ + vkp;

  for (int kt = 0; kt < SEQ; kt += 32) {
    // ---- stage K (32x64) and V (64x32) chunks into LDS ----
    const __bf16* ks = ksrc0 + (size_t)kt * DHEAD;
    const __bf16* vs = vsrc0 + kt;
    async_copy16(ks,     &Ks[kkey * 64 + kdp]);
    async_copy16(ks + 8, &Ks[kkey * 64 + kdp + 8]);
    async_copy16(vs,     &Vs[vd * 32 + vkp]);
    async_copy16(vs + 8, &Vs[vd * 32 + vkp + 8]);
    if (kt + 32 < SEQ) {
      __builtin_prefetch(ks + 32 * DHEAD, 0, 0);
      __builtin_prefetch(vs + 32, 0, 0);
    }
    stage_wait();

    // ---- scores for two 16-key sub-tiles ----
    v8f s[2];
#pragma unroll
    for (int sub = 0; sub < 2; ++sub) {
      const __bf16* krow = &Ks[(sub * 16 + nlo) * 64];
      v16bf bk0 = *(const v16bf*)(krow + half * 16);        // d 0..31
      v16bf bk1 = *(const v16bf*)(krow + 32 + half * 16);   // d 32..63
      v8f c = {};
      c = __builtin_amdgcn_wmma_f32_16x16x32_bf16(false, aq0, false, bk0,
                                                  (short)0, c, false, false);
      c = __builtin_amdgcn_wmma_f32_16x16x32_bf16(false, aq1, false, bk1,
                                                  (short)0, c, false, false);
      const int mk = mptr[kt + sub * 16 + nlo];
#pragma unroll
      for (int r = 0; r < 8; ++r)
        c[r] = (mk == 0) ? NEGVAL : c[r] * 0.125f;   // 1/sqrt(64)
      s[sub] = c;
    }

    // ---- chunk-wide (32 keys) online softmax update ----
    float tm[8];
#pragma unroll
    for (int r = 0; r < 8; ++r) tm[r] = fmaxf(s[0][r], s[1][r]);
#pragma unroll
    for (int d = 1; d < 16; d <<= 1)
#pragma unroll
      for (int r = 0; r < 8; ++r)
        tm[r] = fmaxf(tm[r], __shfl_xor(tm[r], d, 32));
    float alpha[8], rs[8];
#pragma unroll
    for (int r = 0; r < 8; ++r) {
      const float nm = fmaxf(mrow[r], tm[r]);
      alpha[r] = __expf(mrow[r] - nm);
      mrow[r]  = nm;
      const float p0 = __expf(s[0][r] - nm);
      const float p1 = __expf(s[1][r] - nm);
      s[0][r] = p0; s[1][r] = p1;
      rs[r] = p0 + p1;
    }
#pragma unroll
    for (int d = 1; d < 16; d <<= 1)
#pragma unroll
      for (int r = 0; r < 8; ++r) rs[r] += __shfl_xor(rs[r], d, 32);
#pragma unroll
    for (int r = 0; r < 8; ++r) lrow[r] = lrow[r] * alpha[r] + rs[r];
#pragma unroll
    for (int t = 0; t < 4; ++t)
#pragma unroll
      for (int r = 0; r < 8; ++r) acc[t][r] *= alpha[r];

    // ---- stage P (C layout -> row-major 16x32 bf16) in per-wave LDS ----
#pragma unroll
    for (int sub = 0; sub < 2; ++sub)
#pragma unroll
      for (int r = 0; r < 8; ++r)
        pw[(8 * half + r) * 32 + sub * 16 + nlo] = f2bf(s[sub][r]);
    lds_flush_wave();   // wave-private buffer: DScnt drain suffices

    // reload P as A fragment (lane = query row, K = 32 keys of this chunk)
    const __bf16* prow = pw + nlo * 32;
    v16bf pf = cat16(*(const v8bf*)(prow + half * 16),
                     *(const v8bf*)(prow + 8 + half * 16));

#pragma unroll
    for (int t = 0; t < 4; ++t) {
      v16bf bv = *(const v16bf*)&Vs[(t * 16 + nlo) * 32 + half * 16];
      acc[t] = __builtin_amdgcn_wmma_f32_16x16x32_bf16(
          false, pf, false, bv, (short)0, acc[t], false, false);
    }
    __syncthreads();   // all waves done with Ks/Vs before next staging
  }

  // epilogue: divide by softmax denominator, write ctx [B,S,H*64] bf16
#pragma unroll
  for (int t = 0; t < 4; ++t)
#pragma unroll
    for (int r = 0; r < 8; ++r) {
      const float o = acc[t][r] / lrow[r];
      const int srow = q0 + 8 * half + r;
      ctx[(size_t)(b * SEQ + srow) * DMODEL + h * DHEAD + t * 16 + nlo] =
          f2bf(o);
    }
}

// ---------------------------------------------------------------------------
// Kernel 4: output projection.  out = ctx @ Wo^T + bo (fp32), B staged in LDS.
// ---------------------------------------------------------------------------
__global__ __launch_bounds__(128)
void k_oproj(const __bf16* __restrict__ ctx, const __bf16* __restrict__ Wob,
             const float* __restrict__ bo, float* __restrict__ out) {
  __shared__ __bf16 Bs[64 * 32] __attribute__((aligned(32)));

  const int tid  = threadIdx.x;
  const int wave = tid >> 5;
  const int lane = tid & 31;
  const int nlo  = lane & 15;
  const int half = lane >> 4;
  const int row0 = blockIdx.y * 64 + wave * 16;
  const int col0 = blockIdx.x * 64;

  const int scol = tid >> 1;
  const int skp  = (tid & 1) * 16;
  const __bf16* wsrc = Wob + (size_t)(col0 + scol) * DMODEL + skp;

  v8f acc[4] = {{}, {}, {}, {}};
  const __bf16* arow = ctx + (size_t)(row0 + nlo) * DMODEL;

  for (int kb = 0; kb < DMODEL; kb += 32) {
    async_copy16(wsrc + kb,     &Bs[scol * 32 + skp]);
    async_copy16(wsrc + kb + 8, &Bs[scol * 32 + skp + 8]);
    if (kb + 32 < DMODEL) __builtin_prefetch(wsrc + kb + 32, 0, 0);
    stage_wait();

    v16bf a = cat16(*(const v8bf*)(arow + kb + half * 16),
                    *(const v8bf*)(arow + kb + 8 + half * 16));
#pragma unroll
    for (int t = 0; t < 4; ++t) {
      v16bf bf = *(const v16bf*)&Bs[(t * 16 + nlo) * 32 + half * 16];
      acc[t] = __builtin_amdgcn_wmma_f32_16x16x32_bf16(
          false, a, false, bf, (short)0, acc[t], false, false);
    }
    __syncthreads();
  }
#pragma unroll
  for (int t = 0; t < 4; ++t) {
    const int col = col0 + t * 16 + nlo;
    const float bb = bo[col];
#pragma unroll
    for (int r = 0; r < 8; ++r) {
      const int row = row0 + 8 * half + r;
      out[(size_t)row * DMODEL + col] = acc[t][r] + bb;
    }
  }
}

// ---------------------------------------------------------------------------
extern "C" void kernel_launch(void* const* d_in, const int* in_sizes, int n_in,
                              void* d_out, int out_size, void* d_ws,
                              size_t ws_size, hipStream_t stream) {
  const float* query = (const float*)d_in[0];
  const float* key_  = (const float*)d_in[1];
  const float* value = (const float*)d_in[2];
  const int*   mask  = (const int*)d_in[3];
  const float* Wq = (const float*)d_in[4];
  const float* bq = (const float*)d_in[5];
  const float* Wk = (const float*)d_in[6];
  const float* bk = (const float*)d_in[7];
  const float* Wv = (const float*)d_in[8];
  const float* bv = (const float*)d_in[9];
  const float* Wo = (const float*)d_in[10];
  const float* bo = (const float*)d_in[11];

  __bf16* ws  = (__bf16*)d_ws;
  __bf16* Wb  = ws;                              // 4*DD  (Wq,Wk,Wv,Wo bf16)
  __bf16* Qb  = ws + (size_t)4 * DD;             // [B,H,S,64]
  __bf16* Kb  = Qb + (size_t)ND;                 // [B,H,S,64]
  __bf16* Vt  = Kb + (size_t)ND;                 // [B,H,64,S]
  __bf16* ctx = Vt + (size_t)ND;                 // [B,S,768]
  float* out = (float*)d_out;

  k_cvt_w<<<dim3((DD / 4 + 255) / 256, 4), 256, 0, stream>>>(Wq, Wk, Wv, Wo,
                                                             Wb);
  k_proj<<<dim3(DMODEL / 64, NROWS / 64, 3), 128, 0, stream>>>(
      query, key_, value, Wb, bq, bk, bv, Qb, Kb, Vt);
  k_attn<<<dim3(SEQ / 64, BATCH * NHEAD), 128, 0, stream>>>(Qb, Kb, Vt, mask,
                                                            ctx);
  k_oproj<<<dim3(DMODEL / 64, NROWS / 64), 128, 0, stream>>>(
      ctx, Wb + (size_t)3 * DD, bo, out);
}
